// RelationClassifier_58987080843616
// MI455X (gfx1250) — matrix operand order repaired and checked by reading, not compiled, not run on hardware
//
#include <hip/hip_runtime.h>
#include <hip/hip_bf16.h>

// ---------------------------------------------------------------------------
// Problem constants (from the reference)
// ---------------------------------------------------------------------------
#define BB 8
#define NN 64
#define LL 256
#define HH 768
#define EE 1536           // 2*H
#define NHD 12            // heads
#define HD 128            // head dim
#define HL 1024
#define CC 42

typedef __attribute__((ext_vector_type(16))) _Float16 v16h;
typedef __attribute__((ext_vector_type(8)))  _Float16 v8h;
typedef __attribute__((ext_vector_type(8)))  float    v8f;

// ---------------------------------------------------------------------------
// CDNA5 async global->LDS copy (16B per lane) + wait.
// LDS address = low 32 bits of the generic pointer (aperture mapping:
// generic shared ptr = {SHARED_BASE, offset32}, ISA 10.2).
// ---------------------------------------------------------------------------
__device__ inline unsigned lds_off(const void* p) {
    return (unsigned)(unsigned long long)(size_t)p;
}
__device__ inline void async_ld_b128(unsigned lds_addr, const void* gptr) {
    asm volatile("global_load_async_to_lds_b128 %0, %1, off"
                 :: "v"(lds_addr), "v"((unsigned long long)(size_t)gptr)
                 : "memory");
}
__device__ inline void wait_async0() {
    asm volatile("s_wait_asynccnt 0x0" ::: "memory");
}

// ---------------------------------------------------------------------------
// WMMA fragment loaders (layouts per CDNA5 ISA 7.12.2, wave32)
// A 16x32 f16: lanes 0-15 -> row M=lane, halves = K{0..7,16..23};
//              lanes 16-31 -> row M=lane-16, halves = K{8..15,24..31}
// B 32x16 f16 staged as [N][K] in LDS: lane n=(lane&15), halves =
//              K{0..7,16..23} (lanes 0-15) / K{8..15,24..31} (lanes 16-31)
// C/D 16x16 f32: VGPR r -> M = r + 8*(lane>>4), N = lane&15
// ---------------------------------------------------------------------------
__device__ inline v16h ld_afrag(const _Float16* s, int ld, int m0, int k0, int lane) {
    int m  = m0 + (lane & 15);
    int ko = k0 + ((lane >> 4) << 3);
    const _Float16* p = s + m * ld + ko;
    v8h lo = *(const v8h*)(p);
    v8h hi = *(const v8h*)(p + 16);
    return __builtin_shufflevector(lo, hi, 0,1,2,3,4,5,6,7,8,9,10,11,12,13,14,15);
}

__device__ inline v16h ld_bfrag(const _Float16* s, int ld, int n0, int k0, int lane) {
    int n  = n0 + (lane & 15);
    int kk = k0 + ((lane >> 4) << 3);
    const _Float16* p = s + n * ld + kk;
    v8h lo = *(const v8h*)(p);
    v8h hi = *(const v8h*)(p + 16);
    return __builtin_shufflevector(lo, hi, 0,1,2,3,4,5,6,7,8,9,10,11,12,13,14,15);
}

// ---------------------------------------------------------------------------
// 1) Gather: cat[b][n] = concat(hidden[b][n][i0], hidden[b][n][i1]),
//    written both f32 (for the avg stage) and f16 (GEMM A operand).
// ---------------------------------------------------------------------------
__global__ void gather_cat(const float* __restrict__ hidden,
                           const int* __restrict__ inds,
                           float* __restrict__ cat,
                           _Float16* __restrict__ cat_h) {
    int bn = blockIdx.x;                       // b*64+n
    int i0 = inds[bn * 2 + 0];
    int i1 = inds[bn * 2 + 1];
    const float* h0 = hidden + ((size_t)bn * LL + i0) * HH;
    const float* h1 = hidden + ((size_t)bn * LL + i1) * HH;
    float* c = cat + (size_t)bn * EE;
    _Float16* ch = cat_h + (size_t)bn * EE;
    for (int e = threadIdx.x; e < HH; e += 256) {
        float a = h0[e], b = h1[e];
        c[e]       = a;
        c[HH + e]  = b;
        ch[e]      = (_Float16)a;
        ch[HH + e] = (_Float16)b;
    }
}

// ---------------------------------------------------------------------------
// 2) WMMA GEMM: C[M=512][N=1536] = A_f16[512][K] * B_f32 (+bias)
//    A tiles are DMA'd global->LDS (raw f16 bytes, async path).
//    B tiles are converted f32->f16 during staging (weights read once).
//    BT=false: B stored [K][N]; BT=true: B stored [N][K].
//    F16OUT: write f16 (next GEMM's A operand), else f32.
//    Block tile 128x128, BK=32, 8 waves in 4(M) x 2(N), 2x4 tiles per wave.
// ---------------------------------------------------------------------------
#define LDT 40   // LDS row stride in halves (80B, 16B aligned)

template<bool BT, bool BIAS, bool F16OUT>
__global__ void __launch_bounds__(256)
gemm_wmma(const _Float16* __restrict__ A, int lda,
          const float* __restrict__ B, int ldb,
          const float* __restrict__ bias,
          float* __restrict__ Cf, _Float16* __restrict__ Ch,
          int ldc, int K) {
    __shared__ __align__(16) _Float16 sA[128 * LDT];
    __shared__ __align__(16) _Float16 sB[128 * LDT];

    const int tid  = threadIdx.x;
    const int lane = tid & 31;
    const int w    = tid >> 5;
    const int wm   = w & 3;          // 0..3  -> M offset wm*32
    const int wn   = w >> 2;         // 0..1  -> N offset wn*64
    const int m0   = blockIdx.y * 128;
    const int n0   = blockIdx.x * 128;
    const unsigned sA0 = lds_off(sA);

    v8f acc[2][4];
    v8f zero{};
    for (int i = 0; i < 2; ++i)
        for (int j = 0; j < 4; ++j) acc[i][j] = zero;

    for (int k0 = 0; k0 < K; k0 += 32) {
        // ---- async DMA A tile: 128 rows x 64B (4 x b128 per row)
        {
            const _Float16* gA = A + (size_t)m0 * lda + k0;
            for (int c = tid; c < 512; c += 256) {       // 2 async ops per wave
                int r = c >> 2, seg = c & 3;
                async_ld_b128(sA0 + (unsigned)(r * LDT + seg * 8) * 2,
                              gA + (size_t)r * lda + seg * 8);
            }
        }
        // ---- stage B tile as [n][k], f32 -> f16
        if (BT) {
            for (int i = tid; i < 128 * 32; i += 256) {
                int n = i >> 5, c = i & 31;
                sB[n * LDT + c] = (_Float16)B[(size_t)(n0 + n) * ldb + k0 + c];
            }
        } else {
            for (int i = tid; i < 128 * 32; i += 256) {
                int n = i & 127, c = i >> 7;
                sB[n * LDT + c] = (_Float16)B[(size_t)(k0 + c) * ldb + n0 + n];
            }
        }
        // prefetch next B K-tile into L2 while we compute
        if (k0 + 32 < K) {
            if (BT) __builtin_prefetch(&B[(size_t)(n0 + (tid >> 1)) * ldb + k0 + 32], 0, 1);
            else    __builtin_prefetch(&B[(size_t)(k0 + 32 + (tid >> 3)) * ldb + n0], 0, 1);
        }
        wait_async0();       // our wave's DMA landed
        __syncthreads();     // every wave's DMA + ds writes visible

        v16h af[2], bf[4];
        for (int tm = 0; tm < 2; ++tm) af[tm] = ld_afrag(sA, LDT, wm * 32 + tm * 16, 0, lane);
        for (int tn = 0; tn < 4; ++tn) bf[tn] = ld_bfrag(sB, LDT, wn * 64 + tn * 16, 0, lane);
        for (int tm = 0; tm < 2; ++tm)
            for (int tn = 0; tn < 4; ++tn)
                acc[tm][tn] = __builtin_amdgcn_wmma_f32_16x16x32_f16(
                    false, af[tm], false, bf[tn], (short)0, acc[tm][tn], false, false);
        __syncthreads();
    }

    for (int tm = 0; tm < 2; ++tm)
        for (int tn = 0; tn < 4; ++tn)
            for (int r = 0; r < 8; ++r) {
                int row = m0 + wm * 32 + tm * 16 + ((lane >> 4) << 3) + r;
                int col = n0 + wn * 64 + tn * 16 + (lane & 15);
                float vv = acc[tm][tn][r];
                if (BIAS) vv += bias[col];
                if (F16OUT) Ch[(size_t)row * ldc + col] = (_Float16)vv;
                else        Cf[(size_t)row * ldc + col] = vv;
            }
}

// ---------------------------------------------------------------------------
// 3) Attention: one block per (b, head). S = Q K^T / sqrt(128); P = softmax(S);
//    O = P V. Q,K,V are f16 (B,N,E) with head at cols h*128..h*128+127.
//    Q,K tiles DMA'd async; V staged manually (transpose to d-major).
// ---------------------------------------------------------------------------
#define SMQK 17408            // 64 * 136 halves
#define SMVT 18432            // 128 * 72 halves

__global__ void __launch_bounds__(256)
attention_kernel(const _Float16* __restrict__ q, const _Float16* __restrict__ k,
                 const _Float16* __restrict__ v, _Float16* __restrict__ ctx) {
    __shared__ __align__(16) char smem[SMQK + SMQK + SMVT];
    _Float16* sQ  = (_Float16*)smem;                 // 64 x 136
    _Float16* sK  = (_Float16*)(smem + SMQK);        // 64 x 136
    _Float16* sVt = (_Float16*)(smem + 2 * SMQK);    // 128 x 72 (d-major)
    float*    sS  = (float*)smem;                    // 64 x 68 f32, overlays sQ
    _Float16* sP  = (_Float16*)(smem + SMQK);        // 64 x 72, overlays sK

    const int tid = threadIdx.x, lane = tid & 31, w = tid >> 5;
    const int b = blockIdx.x / NHD, h = blockIdx.x % NHD;
    const _Float16* qp = q + (size_t)(b * NN) * EE + h * HD;
    const _Float16* kp = k + (size_t)(b * NN) * EE + h * HD;
    const _Float16* vp = v + (size_t)(b * NN) * EE + h * HD;
    const unsigned sQ0 = lds_off(sQ), sK0 = lds_off(sK);

    // async DMA Q,K: 64 rows x 256B (16 x b128 per row)
    for (int c = tid; c < 1024; c += 256) {
        int n = c >> 4, seg = c & 15;
        unsigned loff = (unsigned)(n * 136 + seg * 8) * 2;
        const size_t goff = (size_t)n * EE + seg * 8;
        async_ld_b128(sQ0 + loff, qp + goff);
        async_ld_b128(sK0 + loff, kp + goff);
    }
    // manual transpose staging of V (f16 copy, d-major)
    for (int i = tid; i < NN * HD; i += 256) {
        int n = i >> 7, d = i & 127;
        sVt[d * 72 + n] = vp[(size_t)n * EE + d];
    }
    wait_async0();
    __syncthreads();

    // ---- S = Q K^T * scale : 4x4 tiles, wave w -> row tile (w&3), col tiles 2*(w>>2)+{0,1}
    {
        const int tm  = w & 3;
        const int tnb = (w >> 2) * 2;
        v8f acc[2]; v8f z{}; acc[0] = z; acc[1] = z;
        for (int k0 = 0; k0 < HD; k0 += 32) {
            v16h af = ld_afrag(sQ, 136, tm * 16, k0, lane);
            for (int j = 0; j < 2; ++j) {
                v16h bf = ld_bfrag(sK, 136, (tnb + j) * 16, k0, lane);
                acc[j] = __builtin_amdgcn_wmma_f32_16x16x32_f16(
                    false, af, false, bf, (short)0, acc[j], false, false);
            }
        }
        __syncthreads();   // everyone done reading sQ/sK before the overlay store
        const float scale = 0.088388347648318447f;   // 1/sqrt(128)
        for (int j = 0; j < 2; ++j)
            for (int r = 0; r < 8; ++r) {
                int row = tm * 16 + ((lane >> 4) << 3) + r;
                int col = (tnb + j) * 16 + (lane & 15);
                sS[row * 68 + col] = acc[j][r] * scale;
            }
    }
    __syncthreads();

    // ---- row softmax (wave w owns rows w*8 .. w*8+7), write f16 P
    for (int rr = 0; rr < 8; ++rr) {
        int r = w * 8 + rr;
        float v0 = sS[r * 68 + lane];
        float v1 = sS[r * 68 + 32 + lane];
        float mx = fmaxf(v0, v1);
        for (int off = 16; off; off >>= 1) mx = fmaxf(mx, __shfl_xor(mx, off, 32));
        float e0 = __expf(v0 - mx), e1 = __expf(v1 - mx);
        float s = e0 + e1;
        for (int off = 16; off; off >>= 1) s += __shfl_xor(s, off, 32);
        float inv = 1.0f / s;
        sP[r * 72 + lane]      = (_Float16)(e0 * inv);
        sP[r * 72 + 32 + lane] = (_Float16)(e1 * inv);
    }
    __syncthreads();

    // ---- O = P V : 4x8 tiles, wave w -> row tile (w&3), col tiles 4*(w>>2)+{0..3}
    {
        const int tm  = w & 3;
        const int tnb = (w >> 2) * 4;
        v8f acc[4]; v8f z{};
        for (int j = 0; j < 4; ++j) acc[j] = z;
        for (int k0 = 0; k0 < NN; k0 += 32) {
            v16h af = ld_afrag(sP, 72, tm * 16, k0, lane);
            for (int j = 0; j < 4; ++j) {
                v16h bf = ld_bfrag(sVt, 72, (tnb + j) * 16, k0, lane);
                acc[j] = __builtin_amdgcn_wmma_f32_16x16x32_f16(
                    false, af, false, bf, (short)0, acc[j], false, false);
            }
        }
        for (int j = 0; j < 4; ++j)
            for (int r = 0; r < 8; ++r) {
                int n = tm * 16 + ((lane >> 4) << 3) + r;
                int d = (tnb + j) * 16 + (lane & 15);
                ctx[(size_t)(b * NN + n) * EE + h * HD + d] = acc[j][r];
            }
    }
}

// ---------------------------------------------------------------------------
// 4) avg[b][e] = (1/64) * sum_n attn_out[b][n][e] * cat[b][n][e]
// ---------------------------------------------------------------------------
__global__ void avg_kernel(const float* __restrict__ attn_out,
                           const float* __restrict__ cat,
                           float* __restrict__ avg) {
    int e = blockIdx.x * 256 + threadIdx.x;
    int b = blockIdx.y;
    float s = 0.f;
    for (int n = 0; n < NN; ++n) {
        size_t idx = (size_t)(b * NN + n) * EE + e;
        s += attn_out[idx] * cat[idx];
    }
    avg[b * EE + e] = s * (1.0f / (float)NN);
}

// ---------------------------------------------------------------------------
// 5) LSTM step: gates GEMV (one wave per output row, both directions),
//    then elementwise state update.
// ---------------------------------------------------------------------------
__global__ void __launch_bounds__(256)
lstm_gates(const float* __restrict__ avg, const float* __restrict__ hstate,
           const float* __restrict__ Wih_f, const float* __restrict__ Whh_f,
           const float* __restrict__ bih_f, const float* __restrict__ bhh_f,
           const float* __restrict__ Wih_b, const float* __restrict__ Whh_b,
           const float* __restrict__ bih_b, const float* __restrict__ bhh_b,
           float* __restrict__ g, int t, int first) {
    int gid  = blockIdx.x * 8 + (threadIdx.x >> 5);  // 0..8191
    int lane = threadIdx.x & 31;
    int dir  = gid >> 12;                            // 0 fwd, 1 bwd
    int row  = gid & 4095;
    const float* x   = avg + (dir == 0 ? t : 7 - t) * EE;
    const float* Wih = dir ? Wih_b : Wih_f;
    const float* Whh = dir ? Whh_b : Whh_f;
    const float* bih = dir ? bih_b : bih_f;
    const float* bhh = dir ? bhh_b : bhh_f;
    const float* hh  = hstate + dir * HL;

    float s = 0.f;
    const float* wr = Wih + (size_t)row * EE;
    for (int j = lane; j < EE; j += 32) s += x[j] * wr[j];
    if (!first) {
        const float* hr = Whh + (size_t)row * HL;
        for (int j = lane; j < HL; j += 32) s += hh[j] * hr[j];
    }
    for (int off = 16; off; off >>= 1) s += __shfl_xor(s, off, 32);
    if (lane == 0) g[dir * 4 * HL + row] = s + bih[row] + bhh[row];
}

__global__ void lstm_update(const float* __restrict__ g, float* __restrict__ hstate,
                            float* __restrict__ cstate, float* __restrict__ lstm_out,
                            int t, int first) {
    int id = blockIdx.x * 256 + threadIdx.x;   // 0..2047
    if (id >= 2 * HL) return;
    int dir = id >> 10, j = id & (HL - 1);
    const float* gd = g + dir * 4 * HL;
    float ig = 1.f / (1.f + __expf(-gd[j]));
    float fg = 1.f / (1.f + __expf(-gd[HL + j]));
    float gg = tanhf(gd[2 * HL + j]);
    float og = 1.f / (1.f + __expf(-gd[3 * HL + j]));
    float c  = first ? 0.f : cstate[dir * HL + j];
    c = fg * c + ig * gg;
    float h = og * tanhf(c);
    cstate[dir * HL + j] = c;
    hstate[dir * HL + j] = h;
    int rowt = (dir == 0) ? t : 7 - t;
    lstm_out[rowt * 2 * HL + dir * HL + j] = h;
}

// ---------------------------------------------------------------------------
// 6) logits + softmax over classes: out[8][42]
// ---------------------------------------------------------------------------
__global__ void logits_softmax(const float* __restrict__ lstm_out,
                               const float* __restrict__ out_w,
                               const float* __restrict__ out_b,
                               float* __restrict__ out) {
    __shared__ float sl[CC];
    int row = blockIdx.x, c = threadIdx.x;
    if (c < CC) {
        float s = out_b[c];
        const float* lr = lstm_out + row * 2 * HL;
        const float* wr = out_w + (size_t)c * 2 * HL;
        for (int j = 0; j < 2 * HL; ++j) s += lr[j] * wr[j];
        sl[c] = s;
    }
    __syncthreads();
    if (c < CC) {
        float mx = -1e30f;
        for (int j = 0; j < CC; ++j) mx = fmaxf(mx, sl[j]);
        float sum = 0.f;
        for (int j = 0; j < CC; ++j) sum += __expf(sl[j] - mx);
        out[row * CC + c] = __expf(sl[c] - mx) / sum;
    }
}

// ---------------------------------------------------------------------------
// Launch
// ---------------------------------------------------------------------------
extern "C" void kernel_launch(void* const* d_in, const int* in_sizes, int n_in,
                              void* d_out, int out_size, void* d_ws, size_t ws_size,
                              hipStream_t stream) {
    const float* hidden     = (const float*)d_in[0];
    const int*   inds       = (const int*)  d_in[1];
    const float* Wkey       = (const float*)d_in[2];
    const float* Wque       = (const float*)d_in[3];
    const float* Wval       = (const float*)d_in[4];
    const float* in_proj_w  = (const float*)d_in[5];
    const float* in_proj_b  = (const float*)d_in[6];
    const float* out_proj_w = (const float*)d_in[7];
    const float* out_proj_b = (const float*)d_in[8];
    const float* Wih_f      = (const float*)d_in[9];
    const float* Whh_f      = (const float*)d_in[10];
    const float* bih_f      = (const float*)d_in[11];
    const float* bhh_f      = (const float*)d_in[12];
    const float* Wih_b      = (const float*)d_in[13];
    const float* Whh_b      = (const float*)d_in[14];
    const float* bih_b      = (const float*)d_in[15];
    const float* bhh_b      = (const float*)d_in[16];
    const float* out_w      = (const float*)d_in[17];
    const float* out_b      = (const float*)d_in[18];

    const size_t SZ = (size_t)BB * NN * EE;     // 786432 activations per tensor
    char* p = (char*)d_ws;
    float*    cat    = (float*)p;     p += SZ * sizeof(float);
    float*    aout   = (float*)p;     p += SZ * sizeof(float);
    float*    avg    = (float*)p;     p += (size_t)BB * EE * sizeof(float);
    float*    gbuf   = (float*)p;     p += (size_t)2 * 4 * HL * sizeof(float);
    float*    hst    = (float*)p;     p += (size_t)2 * HL * sizeof(float);
    float*    cst    = (float*)p;     p += (size_t)2 * HL * sizeof(float);
    float*    lout   = (float*)p;     p += (size_t)BB * 2 * HL * sizeof(float);
    _Float16* cat_h  = (_Float16*)p;  p += SZ * sizeof(_Float16);
    _Float16* keys_h = (_Float16*)p;  p += SZ * sizeof(_Float16);
    _Float16* ques_h = (_Float16*)p;  p += SZ * sizeof(_Float16);
    _Float16* vals_h = (_Float16*)p;  p += SZ * sizeof(_Float16);
    _Float16* qb_h   = (_Float16*)p;  p += SZ * sizeof(_Float16);
    _Float16* kb_h   = (_Float16*)p;  p += SZ * sizeof(_Float16);
    _Float16* vb_h   = (_Float16*)p;  p += SZ * sizeof(_Float16);
    _Float16* ctx_h  = (_Float16*)p;  p += SZ * sizeof(_Float16);

    // 1) gather (f32 + f16)
    gather_cat<<<BB * NN, 256, 0, stream>>>(hidden, inds, cat, cat_h);

    // 2) keys/queries/values = cat @ W*   (B stored [K][N]); f16 outputs
    dim3 gg(EE / 128, (BB * NN) / 128);   // (12, 4)
    gemm_wmma<false, false, true><<<gg, 256, 0, stream>>>(cat_h, EE, Wkey, EE, nullptr, nullptr, keys_h, EE, EE);
    gemm_wmma<false, false, true><<<gg, 256, 0, stream>>>(cat_h, EE, Wque, EE, nullptr, nullptr, ques_h, EE, EE);
    gemm_wmma<false, false, true><<<gg, 256, 0, stream>>>(cat_h, EE, Wval, EE, nullptr, nullptr, vals_h, EE, EE);

    // 3) q/k/v projections (note: _mha(q_in=keys, k_in=queries, v_in=values))
    gemm_wmma<true, true, true><<<gg, 256, 0, stream>>>(keys_h, EE, in_proj_w,                  EE, in_proj_b,        nullptr, qb_h, EE, EE);
    gemm_wmma<true, true, true><<<gg, 256, 0, stream>>>(ques_h, EE, in_proj_w + (size_t)EE*EE,  EE, in_proj_b + EE,   nullptr, kb_h, EE, EE);
    gemm_wmma<true, true, true><<<gg, 256, 0, stream>>>(vals_h, EE, in_proj_w + (size_t)2*EE*EE,EE, in_proj_b + 2*EE, nullptr, vb_h, EE, EE);

    // 4) attention per (b, head) -> ctx f16
    attention_kernel<<<BB * NHD, 256, 0, stream>>>(qb_h, kb_h, vb_h, ctx_h);

    // 5) output projection (f32 out)
    gemm_wmma<true, true, false><<<gg, 256, 0, stream>>>(ctx_h, EE, out_proj_w, EE, out_proj_b, aout, nullptr, EE, EE);

    // 6) weighted average over N
    avg_kernel<<<dim3(EE / 256, BB), 256, 0, stream>>>(aout, cat, avg);

    // 7) bidirectional LSTM, 8 sequential steps
    for (int t = 0; t < BB; ++t) {
        lstm_gates<<<1024, 256, 0, stream>>>(avg, hst,
                                             Wih_f, Whh_f, bih_f, bhh_f,
                                             Wih_b, Whh_b, bih_b, bhh_b,
                                             gbuf, t, t == 0);
        lstm_update<<<8, 256, 0, stream>>>(gbuf, hst, cst, lout, t, t == 0);
    }

    // 8) classifier + softmax
    logits_softmax<<<BB, 64, 0, stream>>>(lout, out_w, out_b, (float*)d_out);
}
